// DGCN_batch_29454885716514
// MI455X (gfx1250) — compile-verified, matrix operand order checked by reading, not run on hardware
//
#include <hip/hip_runtime.h>

typedef __bf16 bf16;
typedef __attribute__((ext_vector_type(16))) bf16  v16bf;
typedef __attribute__((ext_vector_type(8)))  bf16  v8bf;
typedef __attribute__((ext_vector_type(8)))  float v8f;
typedef __attribute__((ext_vector_type(4)))  unsigned int u32x4;
typedef __attribute__((ext_vector_type(8)))  int i32x8;
typedef __attribute__((ext_vector_type(4)))  int i32x4;

union BFrag { v16bf v; v8bf h[2]; };

#if defined(__gfx1250__) && __has_builtin(__builtin_amdgcn_tensor_load_to_lds) && \
    __has_builtin(__builtin_amdgcn_s_wait_tensorcnt)
#define HAVE_TDM 1
#else
#define HAVE_TDM 0
#endif

__device__ __forceinline__ void atomic_add_f32(float* p, float v) {
  __hip_atomic_fetch_add(p, v, __ATOMIC_RELAXED, __HIP_MEMORY_SCOPE_AGENT);
}

__device__ __forceinline__ unsigned short f32_to_bf16(float f) {
  unsigned int u = __float_as_uint(f);
  u += 0x7FFFu + ((u >> 16) & 1u);          // round-to-nearest-even
  return (unsigned short)(u >> 16);
}

// ---------------- small elementwise kernels ----------------
__global__ void k_fill(float* __restrict__ p, int n, float v) {
  int i = blockIdx.x * blockDim.x + threadIdx.x;
  if (i < n) p[i] = v;
}
__global__ void k_rsqrt_ip(float* __restrict__ p, int n) {
  int i = blockIdx.x * blockDim.x + threadIdx.x;
  if (i < n) p[i] = rsqrtf(p[i]);           // deg >= 1 (self-loop) so always valid
}
__global__ void k_relu(float* __restrict__ p, int n) {
  int i = blockIdx.x * blockDim.x + threadIdx.x;
  if (i < n) p[i] = fmaxf(p[i], 0.0f);
}
__global__ void k_f32_to_bf16(const float* __restrict__ in, unsigned short* __restrict__ out, int n) {
  int i = blockIdx.x * blockDim.x + threadIdx.x;
  if (i < n) out[i] = f32_to_bf16(in[i]);
}

// ---------------- graph normalization ----------------
__global__ void k_deg(const int* __restrict__ dst, const float* __restrict__ w,
                      float* __restrict__ deg, int E, int has_w) {
  int e = blockIdx.x * blockDim.x + threadIdx.x;
  if (e < E) atomic_add_f32(&deg[dst[e]], has_w ? w[e] : 1.0f);
}
__global__ void k_norm(const int* __restrict__ src, const int* __restrict__ dst,
                       const float* __restrict__ w, const float* __restrict__ dinv,
                       float* __restrict__ nrm, int E, int has_w) {
  int e = blockIdx.x * blockDim.x + threadIdx.x;
  if (e < E) {
    float ww = has_w ? w[e] : 1.0f;
    nrm[e] = dinv[src[e]] * ww * dinv[dst[e]];
  }
}

// ---------------- WMMA bf16 GEMM: H[M,Nout] = A[M,K] * Wt[Nout,K]^T ----------------
// grid.x = 625 (5 waves/block * 625 = 3125 = M/16 row tiles, exact -> EXEC always full)
// grid.y = Nout/64. The 64-row W slab (contiguous 64*K bf16) is staged in LDS by the
// Tensor Data Mover with pad_enable: 4 DWORDs of pad every 64 DWORDs, i.e. LDS byte
// address of linear element g is 2*g + 16*(g>>7). K is a multiple of 128 elements,
// so per-row stride rowB = 2K + 16*(K/128) bytes and lane strides stay bank-friendly.
__launch_bounds__(160)
__global__ void k_gemm_bf16(const bf16* __restrict__ A, const bf16* __restrict__ Wt,
                            float* __restrict__ H, int K, int Nout) {
  extern __shared__ char lw[];
  const int colblk = blockIdx.y * 64;
  const int rowB = (K << 1) + ((K >> 7) << 4);    // padded bytes per B row in LDS

#if HAVE_TDM
  if ((threadIdx.x >> 5) == 0) {                   // wave 0 issues the TDM descriptor
    unsigned long long ga = (unsigned long long)(const void*)(Wt + (size_t)colblk * K);
    unsigned lds = (unsigned)(unsigned long long)(void*)lw;
    unsigned dim0 = (unsigned)(64 * K);            // 1-D tile, <= 24576 elements
    u32x4 g0;
    g0.x = 1u;                                     // count=1, user descriptor
    g0.y = lds;                                    // lds_addr (bytes)
    g0.z = (unsigned)ga;                           // global_addr[31:0]
    g0.w = (unsigned)((ga >> 32) & 0x01FFFFFFull) | (2u << 30);  // addr[56:32] | type=2
    i32x8 g1;
    g1[0] = (int)((1u << 16) |                     // data_size = 2 bytes
                  (1u << 20) |                     // pad_enable
                  (5u << 22) |                     // pad_interval: every 64 DWORDs
                  (3u << 25));                     // pad_amount: 4 DWORDs (16B)
    g1[1] = (int)((dim0 & 0xFFFFu) << 16);         // tensor_dim0[15:0]
    g1[2] = (int)(((dim0 >> 16) & 0xFFFFu) | (1u << 16));  // tensor_dim0[31:16] | tensor_dim1=1
    g1[3] = (int)(dim0 << 16);                     // tile_dim0 = 64*K
    g1[4] = 0;                                     // tile_dim1/2 unused
    g1[5] = (int)dim0;                             // tensor_dim0_stride
    g1[6] = 0;
    g1[7] = 0;
    i32x4 gz = {0, 0, 0, 0};
#if __has_include(<hip/amd_detail/amd_gfx1250_TDM.h>)
    i32x8 gz8 = {0, 0, 0, 0, 0, 0, 0, 0};         // 6-arg toolchain variant
    __builtin_amdgcn_tensor_load_to_lds(g0, g1, gz, gz, gz8, 0);
#else
    __builtin_amdgcn_tensor_load_to_lds(g0, g1, gz, gz, 0);  // ROCm 7.2: 5 args
#endif
    __builtin_amdgcn_s_wait_tensorcnt(0);
  }
  __syncthreads();
#else
  // fallback: cooperative copy producing the identical padded layout
  for (int c = threadIdx.x; c < 8 * K; c += blockDim.x) {   // 64*K/8 16B chunks
    int g = c << 3;
    *(v8bf*)(lw + ((size_t)g << 1) + ((g >> 7) << 4)) =
        *(const v8bf*)&Wt[(size_t)colblk * K + g];
  }
  __syncthreads();
#endif

  const int wave = threadIdx.x >> 5;
  const int lane = threadIdx.x & 31;
  const int tile = blockIdx.x * 5 + wave;   // 0..3124, always valid -> EXEC all ones
  const int m    = lane & 15;               // row within A tile / col within B tile
  const int khi  = lane >> 4;               // K-half selector (ISA 16-bit layout)

  const bf16* arow = A + (size_t)(tile * 16 + m) * K + khi * 8;
  const char* lb = lw + (size_t)m * rowB;   // this lane's B row base (per 16-col tile +16 rows)
  v8f acc[4] = {};

  for (int k0 = 0; k0 < K; k0 += 32) {
    BFrag a;
    a.h[0] = *(const v8bf*)(arow + k0);
    a.h[1] = *(const v8bf*)(arow + k0 + 16);
    const int kk = k0 + (khi << 3);
    const int o0 = (kk << 1) + ((kk >> 7) << 4);
    const int o1 = ((kk + 16) << 1) + (((kk + 16) >> 7) << 4);
#pragma unroll
    for (int t = 0; t < 4; ++t) {
      BFrag b;
      const char* brow = lb + (size_t)t * 16 * rowB;
      b.h[0] = *(const v8bf*)(brow + o0);
      b.h[1] = *(const v8bf*)(brow + o1);
      acc[t] = __builtin_amdgcn_wmma_f32_16x16x32_bf16(
          false, a.v, false, b.v, (short)0, acc[t], false, false);
    }
  }

  const int row0 = tile * 16 + khi * 8;     // C layout: VGPR r -> row r + 8*khi, col = lane&15
#pragma unroll
  for (int t = 0; t < 4; ++t) {
    float* o = H + (size_t)row0 * Nout + colblk + t * 16 + m;
#pragma unroll
    for (int r = 0; r < 8; ++r) o[(size_t)r * Nout] = acc[t][r];
  }
}

// ---------------- propagate: init (bias + self-loop), then edge scatter ----------------
__global__ void k_init_out(const float* __restrict__ h, const float* __restrict__ dinv,
                           const float* __restrict__ b, float* __restrict__ xout,
                           int N, int dshift, int ld, int coloff) {
  int idx = blockIdx.x * blockDim.x + threadIdx.x;
  if (idx >= (N << dshift)) return;
  int d = 1 << dshift;
  int i = idx >> dshift, f = idx & (d - 1);
  float dv = dinv[i];
  xout[(size_t)i * ld + coloff + f] = b[f] + dv * dv * h[idx];
}

__global__ void k_scatter(const int* __restrict__ src, const int* __restrict__ dst,
                          const float* __restrict__ nrm, const float* __restrict__ h,
                          float* __restrict__ xout, int E, int vshift, int ld, int coloff) {
  long long idx = (long long)blockIdx.x * blockDim.x + threadIdx.x;
  if (idx >= ((long long)E << vshift)) return;
  int e  = (int)(idx >> vshift);
  int f4 = (int)(idx & ((1 << vshift) - 1));
  int s = src[e], q = dst[e];
  float w = nrm[e];
  int d = 4 << vshift;
  const float4 hv = *(const float4*)&h[(size_t)s * d + f4 * 4];
  float* o = &xout[(size_t)q * ld + coloff + f4 * 4];
  atomic_add_f32(o + 0, w * hv.x);
  atomic_add_f32(o + 1, w * hv.y);
  atomic_add_f32(o + 2, w * hv.z);
  atomic_add_f32(o + 3, w * hv.w);
}

// ---------------- head: y = x @ Wc^T + bc, then log_softmax (wave per node) ----------------
__global__ void k_head(const float* __restrict__ x, const float* __restrict__ Wc,
                       const float* __restrict__ bc, float* __restrict__ y, int N) {
  int t = blockIdx.x * blockDim.x + threadIdx.x;
  int node = t >> 5, lane = t & 31;
  if (node >= N) return;
  const float* row = x + (size_t)node * 192;
  float s0 = 0.f, s1 = 0.f, s2 = 0.f, s3 = 0.f;
  for (int f = lane; f < 192; f += 32) {
    float v = row[f];
    s0 = fmaf(v, Wc[f], s0);
    s1 = fmaf(v, Wc[192 + f], s1);
    s2 = fmaf(v, Wc[384 + f], s2);
    s3 = fmaf(v, Wc[576 + f], s3);
  }
  for (int o = 16; o > 0; o >>= 1) {
    s0 += __shfl_down(s0, o);
    s1 += __shfl_down(s1, o);
    s2 += __shfl_down(s2, o);
    s3 += __shfl_down(s3, o);
  }
  if (lane == 0) {
    float y0 = s0 + bc[0], y1 = s1 + bc[1], y2 = s2 + bc[2], y3 = s3 + bc[3];
    float mx = fmaxf(fmaxf(y0, y1), fmaxf(y2, y3));
    float e0 = __expf(y0 - mx), e1 = __expf(y1 - mx), e2 = __expf(y2 - mx), e3 = __expf(y3 - mx);
    float lse = __logf(e0 + e1 + e2 + e3);
    float* o = y + (size_t)node * 4;
    o[0] = y0 - mx - lse; o[1] = y1 - mx - lse; o[2] = y2 - mx - lse; o[3] = y3 - mx - lse;
  }
}

// ---------------- host launcher ----------------
extern "C" void kernel_launch(void* const* d_in, const int* in_sizes, int n_in,
                              void* d_out, int out_size, void* d_ws, size_t ws_size,
                              hipStream_t stream) {
  const int N = in_sizes[0] / 128;          // 50000
  const int E = in_sizes[1] / 2;            // 800000

  const float* x    = (const float*)d_in[0];
  const int*   ei   = (const int*)d_in[1];
  const int*   ein  = (const int*)d_in[2];
  const int*   eout = (const int*)d_in[3];
  const float* in_w  = (const float*)d_in[4];
  const float* out_w = (const float*)d_in[5];
  const float* W0 = (const float*)d_in[6];
  const float* W1 = (const float*)d_in[7];
  const float* W2 = (const float*)d_in[8];
  const float* b0 = (const float*)d_in[9];
  const float* b1 = (const float*)d_in[10];
  const float* b2 = (const float*)d_in[11];
  const float* Wc = (const float*)d_in[12];
  const float* bc = (const float*)d_in[13];

  const int* srcs[3] = { ei, ein, eout };
  const int* dsts[3] = { ei + E, ein + E, eout + E };
  const float* ews[3] = { nullptr, in_w, out_w };

  // workspace carve-out (256B aligned)
  char* wsp = (char*)d_ws;
  size_t off = 0;
  auto alloc = [&](size_t bytes) -> void* {
    void* p = wsp + off;
    off += (bytes + 255) & ~(size_t)255;
    return p;
  };
  float*          dinv = (float*)alloc((size_t)3 * N * 4);        // deg -> dinv in place
  float*          nrm  = (float*)alloc((size_t)3 * E * 4);
  unsigned short* xbf  = (unsigned short*)alloc((size_t)N * 384 * 2);
  unsigned short* wbf  = (unsigned short*)alloc((size_t)128 * 384 * 2);
  float*          hbuf = (float*)alloc((size_t)N * 128 * 4);
  float*          xbuf = (float*)alloc((size_t)N * 384 * 4);

  const int B = 256;
  auto blk = [&](long long n) { return (unsigned)((n + B - 1) / B); };

  // graph norms: deg=1 (self-loop) + scatter w; dinv = rsqrt(deg); edge norm
  k_fill<<<blk(3LL * N), B, 0, stream>>>(dinv, 3 * N, 1.0f);
  for (int g = 0; g < 3; ++g)
    k_deg<<<blk(E), B, 0, stream>>>(dsts[g], ews[g], dinv + (size_t)g * N, E, g > 0);
  k_rsqrt_ip<<<blk(3LL * N), B, 0, stream>>>(dinv, 3 * N);
  for (int g = 0; g < 3; ++g)
    k_norm<<<blk(E), B, 0, stream>>>(srcs[g], dsts[g], ews[g], dinv + (size_t)g * N,
                                     nrm + (size_t)g * E, E, g > 0);

  struct LayerCfg { const float* W; const float* b; int K; int Dout; };
  LayerCfg L[3] = { {W0, b0, 128, 128}, {W1, b1, 384, 128}, {W2, b2, 384, 64} };

  const float* xin = x;
  for (int l = 0; l < 3; ++l) {
    const int K = L[l].K, Dout = L[l].Dout;
    k_f32_to_bf16<<<blk((long long)N * K), B, 0, stream>>>(xin, xbf, N * K);
    k_f32_to_bf16<<<blk((long long)Dout * K), B, 0, stream>>>(L[l].W, wbf, Dout * K);

    dim3 gg(625, Dout / 64);
    size_t shm = (size_t)64 * ((K << 1) + ((K >> 7) << 4));   // padded slab bytes
    k_gemm_bf16<<<gg, 160, shm, stream>>>((const bf16*)xbf, (const bf16*)wbf, hbuf, K, Dout);

    float* xout = (l == 2) ? (float*)d_out : xbuf;
    const int ld = 3 * Dout;
    const int dshift = (Dout == 128) ? 7 : 6;
    const int vshift = dshift - 2;
    for (int g = 0; g < 3; ++g) {
      k_init_out<<<blk((long long)N << dshift), B, 0, stream>>>(
          hbuf, dinv + (size_t)g * N, L[l].b, xout, N, dshift, ld, g * Dout);
      k_scatter<<<blk((long long)E << vshift), B, 0, stream>>>(
          srcs[g], dsts[g], nrm + (size_t)g * E, hbuf, xout, E, vshift, ld, g * Dout);
    }
    k_relu<<<blk((long long)N * ld), B, 0, stream>>>(xout, N * ld);
    xin = xout;
  }

  // head: reads final features from d_out[0 : N*192], writes log-probs after them
  k_head<<<blk((long long)N * 32), B, 0, stream>>>(
      (const float*)d_out, Wc, bc, (float*)d_out + (size_t)N * 192, N);
}
